// MambaTTS_82145544503485
// MI455X (gfx1250) — matrix operand order repaired and checked by reading, not compiled, not run on hardware
//
#include <hip/hip_runtime.h>
#include <cstdint>
#include <cstddef>

// ---------------------------------------------------------------------------
// Model constants (from the reference)
// ---------------------------------------------------------------------------
#define D_MODEL 1024
#define D_INNER 2048
#define D_STATE 16
#define DT_RANK 64
#define D_CONV  4
#define N_HEADS 4
#define D_HEAD  256
#define BATCH   8
#define L_TXT   256
#define L_MEL   1024
#define N_ENC   8
#define N_DEC   4
#define M_ENC   (BATCH * L_TXT)   // 2048
#define M_DEC   (BATCH * L_MEL)   // 8192
#define DBL_LD  128               // padded width of the xproj output (96 -> 128)
#define MEL_KP  128               // padded mel feature dim (80 -> 128, keeps K%64==0)

typedef __attribute__((ext_vector_type(16))) __bf16 v16bf;
typedef __attribute__((ext_vector_type(8)))  float  v8f;
typedef __attribute__((ext_vector_type(4)))  float  v4f;

// ---------------------------------------------------------------------------
// Small device helpers
// ---------------------------------------------------------------------------
__device__ __forceinline__ float silu_f(float x) {
    return x / (1.0f + __expf(-x));
}

// A fragment: 16x32 bf16.  Lane's 16 K-values are contiguous at k0+16*hl
// (same k order as the B fragments).
__device__ __forceinline__ v16bf load_a_frag(const float* __restrict__ Arow,
                                             int k0, int hl) {
    const v4f* pa  = (const v4f*)(Arow + k0 + 16 * hl);
    v4f a0 = pa[0], a1 = pa[1];
    v4f a2 = pa[2], a3 = pa[3];
    v16bf a;
    #pragma unroll
    for (int e = 0; e < 4; ++e) {
        a[e]      = (__bf16)a0[e];
        a[e + 4]  = (__bf16)a1[e];
        a[e + 8]  = (__bf16)a2[e];
        a[e + 12] = (__bf16)a3[e];
    }
    return a;
}

// B fragment, bf16 [N,K]: lane's 16 K-values are contiguous (32B -> 2x b128)
__device__ __forceinline__ v16bf load_b_frag_h(const __bf16* __restrict__ Bh,
                                               long ncol, int ldb, int k0, int hl) {
    return *(const v16bf*)(Bh + ncol * ldb + k0 + 16 * hl);
}

// B fragment, f32 [N,K]
__device__ __forceinline__ v16bf load_b_frag_f(const float* __restrict__ Bf,
                                               long ncol, int ldb, int k0, int hl) {
    const v4f* pb = (const v4f*)(Bf + ncol * ldb + k0 + 16 * hl);
    v4f b0 = pb[0], b1 = pb[1], b2 = pb[2], b3 = pb[3];
    v16bf b;
    #pragma unroll
    for (int e = 0; e < 4; ++e) {
        b[e]      = (__bf16)b0[e];
        b[e + 4]  = (__bf16)b1[e];
        b[e + 8]  = (__bf16)b2[e];
        b[e + 12] = (__bf16)b3[e];
    }
    return b;
}

// B fragment, f32 [K,N] (column access, scalar loads)
__device__ __forceinline__ v16bf load_b_frag_t(const float* __restrict__ Bf,
                                               int ncol, int ldb, int k0, int hl) {
    const float* bp = Bf + (long)(k0 + 16 * hl) * ldb + ncol;
    v16bf b;
    #pragma unroll
    for (int e = 0; e < 16; ++e)
        b[e] = (__bf16)bp[(long)e * ldb];
    return b;
}

// ---------------------------------------------------------------------------
// f32 -> bf16 conversion (weights, once per launch)
// ---------------------------------------------------------------------------
__global__ void k_cvt_bf16(const float* __restrict__ s, __bf16* __restrict__ d, long n) {
    long i = (long)blockIdx.x * blockDim.x + threadIdx.x;
    if (i < n) d[i] = (__bf16)s[i];
}

// f32 [rows,Ks] -> bf16 [rows,Kd] with zero right-padding
__global__ void k_cvt_pad_bf16(const float* __restrict__ s, __bf16* __restrict__ d,
                               long rows, int Ks, int Kd) {
    long i = (long)blockIdx.x * blockDim.x + threadIdx.x;
    if (i >= rows * Kd) return;
    long r = i / Kd;
    int  k = (int)(i - r * Kd);
    d[i] = (k < Ks) ? (__bf16)s[r * Ks + k] : (__bf16)0.f;
}

// f32 [rows,Ks] -> f32 [rows,Kd] with zero right-padding
__global__ void k_pad_f32(const float* __restrict__ s, float* __restrict__ d,
                          long rows, int Ks, int Kd) {
    long i = (long)blockIdx.x * blockDim.x + threadIdx.x;
    if (i >= rows * Kd) return;
    long r = i / Kd;
    int  k = (int)(i - r * Kd);
    d[i] = (k < Ks) ? s[r * Ks + k] : 0.f;
}

// ---------------------------------------------------------------------------
// Embedding gather: x[row, j] = emb[ids[row], j]
// ---------------------------------------------------------------------------
__global__ void k_embed(const int* __restrict__ ids, const float* __restrict__ emb,
                        float* __restrict__ x, long rows, int d) {
    long i = (long)blockIdx.x * blockDim.x + threadIdx.x;
    if (i >= rows * d) return;
    long r = i / d;
    int  j = (int)(i - r * d);
    x[i] = emb[(long)ids[r] * d + j];
}

// ---------------------------------------------------------------------------
// RMSNorm (one block per row, blockDim = 256)
// ---------------------------------------------------------------------------
__global__ void k_rmsnorm(const float* __restrict__ x, const float* __restrict__ w,
                          float* __restrict__ o, int d) {
    long row = blockIdx.x;
    int  t   = threadIdx.x;
    const float* xr = x + row * (long)d;
    float s = 0.f;
    for (int j = t; j < d; j += 256) { float v = xr[j]; s += v * v; }
    __shared__ float sm[256];
    sm[t] = s; __syncthreads();
    for (int st = 128; st > 0; st >>= 1) {
        if (t < st) sm[t] += sm[t + st];
        __syncthreads();
    }
    float rs = rsqrtf(sm[0] / (float)d + 1e-5f);
    for (int j = t; j < d; j += 256) o[row * (long)d + j] = xr[j] * rs * w[j];
}

// ---------------------------------------------------------------------------
// LayerNorm (one block per row, blockDim = 256)
// ---------------------------------------------------------------------------
__global__ void k_layernorm(const float* __restrict__ x, const float* __restrict__ w,
                            const float* __restrict__ b, float* __restrict__ o, int d) {
    long row = blockIdx.x;
    int  t   = threadIdx.x;
    const float* xr = x + row * (long)d;
    float s = 0.f, s2 = 0.f;
    for (int j = t; j < d; j += 256) { float v = xr[j]; s += v; s2 += v * v; }
    __shared__ float sm[256], sq[256];
    sm[t] = s; sq[t] = s2; __syncthreads();
    for (int st = 128; st > 0; st >>= 1) {
        if (t < st) { sm[t] += sm[t + st]; sq[t] += sq[t + st]; }
        __syncthreads();
    }
    float mu  = sm[0] / (float)d;
    float var = sq[0] / (float)d - mu * mu;
    float rs  = rsqrtf(var + 1e-5f);
    for (int j = t; j < d; j += 256)
        o[row * (long)d + j] = (xr[j] - mu) * rs * w[j] + b[j];
}

// ---------------------------------------------------------------------------
// Row softmax, width == 256 == blockDim
// ---------------------------------------------------------------------------
__global__ void k_softmax256(float* __restrict__ s) {
    long row = blockIdx.x;
    int  t   = threadIdx.x;
    float* r = s + row * 256;
    float v = r[t];
    __shared__ float sm[256];
    sm[t] = v; __syncthreads();
    for (int st = 128; st > 0; st >>= 1) {
        if (t < st) sm[t] = fmaxf(sm[t], sm[t + st]);
        __syncthreads();
    }
    float mx = sm[0];
    __syncthreads();
    float e = __expf(v - mx);
    sm[t] = e; __syncthreads();
    for (int st = 128; st > 0; st >>= 1) {
        if (t < st) sm[t] += sm[t + st];
        __syncthreads();
    }
    r[t] = e / sm[0];
}

// ---------------------------------------------------------------------------
// Causal depthwise conv1d (width 4, left-pad 3) + bias + SiLU.
// Input = first D_INNER columns of inproj output (ld = 2*D_INNER).
// ---------------------------------------------------------------------------
__global__ void k_conv_silu(const float* __restrict__ ip, const float* __restrict__ cw,
                            const float* __restrict__ cb, float* __restrict__ xc,
                            int Bn, int L) {
    long i = (long)blockIdx.x * blockDim.x + threadIdx.x;
    long tot = (long)Bn * L * D_INNER;
    if (i >= tot) return;
    int  d  = (int)(i % D_INNER);
    long rl = i / D_INNER;
    int  l  = (int)(rl % L);
    long b  = rl / L;
    float acc = cb[d];
    #pragma unroll
    for (int j = 0; j < D_CONV; ++j) {
        int ls = l - (D_CONV - 1) + j;
        float v = (ls >= 0) ? ip[((b * L + ls) * (long)(2 * D_INNER)) + d] : 0.f;
        acc += cw[d * D_CONV + j] * v;
    }
    xc[rl * (long)D_INNER + d] = silu_f(acc);
}

// ---------------------------------------------------------------------------
// Softplus in place
// ---------------------------------------------------------------------------
__global__ void k_softplus(float* __restrict__ x, long n) {
    long i = (long)blockIdx.x * blockDim.x + threadIdx.x;
    if (i >= n) return;
    float v = x[i];
    x[i] = (v > 20.f) ? v : log1pf(__expf(v));
}

// ---------------------------------------------------------------------------
// Selective scan: one thread per (b, d) channel, 16-float state in registers.
// Fuses the `+ xi*D` skip term and the `* silu(res)` gate.
//  dbl layout per row (ld = DBL_LD): [0:64]=dt, [64:80]=B, [80:96]=C
// ---------------------------------------------------------------------------
__global__ void k_scan(const float* __restrict__ dlt, const float* __restrict__ dbl,
                       const float* __restrict__ xc, const float* __restrict__ ip,
                       const float* __restrict__ Alog, const float* __restrict__ Dp,
                       float* __restrict__ ys, int Bn, int L) {
    int i = blockIdx.x * blockDim.x + threadIdx.x;
    if (i >= Bn * D_INNER) return;
    int b = i / D_INNER;
    int d = i % D_INNER;

    float Aa[D_STATE], hst[D_STATE];
    #pragma unroll
    for (int n = 0; n < D_STATE; ++n) {
        Aa[n]  = -__expf(Alog[(long)d * D_STATE + n]);
        hst[n] = 0.f;
    }
    float Dd = Dp[d];

    for (int l = 0; l < L; ++l) {
        long row = (long)b * L + l;
        float dl = dlt[row * D_INNER + d];
        float u  = xc [row * D_INNER + d];
        const float* bc = dbl + row * DBL_LD;
        float acc = 0.f;
        #pragma unroll
        for (int n = 0; n < D_STATE; ++n) {
            float dA = __expf(dl * Aa[n]);
            hst[n] = dA * hst[n] + dl * bc[64 + n] * u;
            acc   += hst[n] * bc[80 + n];
        }
        float r = ip[row * (long)(2 * D_INNER) + D_INNER + d];   // gate (res half)
        ys[row * D_INNER + d] = (acc + u * Dd) * silu_f(r);
    }
}

// ---------------------------------------------------------------------------
// WMMA GEMM:  C[M,N] = alpha * A[M,K] * op(B) + bias[n] + res[m,n]
//   BMODE 0: B is bf16 [N,K]  (weights)
//   BMODE 1: B is f32  [N,K]  (K-matrix for attention scores)
//   BMODE 2: B is f32  [K,N]  (V-matrix for attention context)
//
// Hot-loop contract (caller guarantees, via buffer padding):
//   * M % 64 == 0, grid.y == M/64          -> no m guards
//   * K % 64 == 0                          -> no k guards, K unrolled by 2
//   * B has >= gridDim.x*128 valid rows (BMODE 0/1) resp. full N cols (2)
//     -> no n guards on loads; only the epilogue store checks ncol < N.
// Software pipelining: all 2 A-fragments + 8 B-fragments for a 64-deep K
// slice are issued before the 8 back-to-back WMMAs, so the second group's
// loads overlap the first group's matrix work (loads return in order).
// One wave -> 16x64 strip; block (8 waves) -> 64x128 tile.  EXEC stays
// all-ones at every v_wmma.  A/B: contiguous k = 16*hl + s; C/D per CDNA5 ISA 7.12.2 (wave32).
// ---------------------------------------------------------------------------
template <int BMODE>
__global__ __launch_bounds__(256) void k_gemm(
    const float* __restrict__ A, const void* __restrict__ Bptr,
    const float* __restrict__ bias, const float* __restrict__ res,
    float* __restrict__ C,
    int M, int N, int K, int lda, int ldb, int ldc, int ldres, float alpha,
    int Hbat, long sAb, long sAh, long sBb, long sBh, long sCb, long sCh) {

    int zb = blockIdx.z / Hbat, zh = blockIdx.z % Hbat;
    A += zb * sAb + zh * sAh;
    C += zb * sCb + zh * sCh;
    long boff = zb * sBb + zh * sBh;
    const __bf16* Bh = (const __bf16*)Bptr + boff;
    const float*  Bf = (const float*)Bptr + boff;
    (void)Bh; (void)Bf; (void)M;

    const int lane = threadIdx.x & 31;
    const int wv   = threadIdx.x >> 5;
    const int m0   = blockIdx.y * 64 + (wv >> 1) * 16;
    const int n0   = blockIdx.x * 128 + (wv & 1) * 64;
    const int hl   = lane >> 4;       // lane half: K sub-range select / M+8
    const int l16  = lane & 15;
    const int mrow = m0 + l16;        // A fragment row for this lane

    v8f acc[4] = {};

    const float* Arow = A + (long)mrow * lda;

    for (int k0 = 0; k0 < K; k0 += 64) {
        if (k0 + 64 < K)
            __builtin_prefetch(&Arow[k0 + 64], 0, 3);

        // ---- issue ALL loads for the 64-deep K slice first ----
        v16bf a0 = load_a_frag(Arow, k0, hl);
        v16bf a1 = load_a_frag(Arow, k0 + 32, hl);
        v16bf b0[4], b1[4];
        #pragma unroll
        for (int t = 0; t < 4; ++t) {
            long ncol = n0 + t * 16 + l16;
            if constexpr (BMODE == 0) {
                b0[t] = load_b_frag_h(Bh, ncol, ldb, k0, hl);
                b1[t] = load_b_frag_h(Bh, ncol, ldb, k0 + 32, hl);
            } else if constexpr (BMODE == 1) {
                b0[t] = load_b_frag_f(Bf, ncol, ldb, k0, hl);
                b1[t] = load_b_frag_f(Bf, ncol, ldb, k0 + 32, hl);
            } else {
                b0[t] = load_b_frag_t(Bf, (int)ncol, ldb, k0, hl);
                b1[t] = load_b_frag_t(Bf, (int)ncol, ldb, k0 + 32, hl);
            }
        }

        // ---- 8 back-to-back WMMAs ----
        #pragma unroll
        for (int t = 0; t < 4; ++t)
            acc[t] = __builtin_amdgcn_wmma_f32_16x16x32_bf16(
                false, a0, false, b0[t], (short)0, acc[t], false, false);
        #pragma unroll
        for (int t = 0; t < 4; ++t)
            acc[t] = __builtin_amdgcn_wmma_f32_16x16x32_bf16(
                false, a1, false, b1[t], (short)0, acc[t], false, false);
    }

    // ---- epilogue: C/D layout (VGPR e -> M = e + 8*hl, lane -> N) ----
    #pragma unroll
    for (int t = 0; t < 4; ++t) {
        int ncol = n0 + t * 16 + l16;
        if (ncol < N) {
            #pragma unroll
            for (int e = 0; e < 8; ++e) {
                int m = m0 + 8 * hl + e;
                float vv = acc[t][e] * alpha;
                if (bias) vv += bias[ncol];
                if (res)  vv += res[(long)m * ldres + ncol];
                C[(long)m * ldc + ncol] = vv;
            }
        }
    }
}

// ---------------------------------------------------------------------------
// Host-side helpers
// ---------------------------------------------------------------------------
static void cvt(hipStream_t s, const float* src, __bf16* dst, long n) {
    k_cvt_bf16<<<(unsigned)((n + 255) / 256), 256, 0, s>>>(src, dst, n);
}

static void gemm_w(hipStream_t s, const float* A, const __bf16* W,
                   const float* bias, const float* res, float* C,
                   int M, int N, int K, int lda, int ldb, int ldc, int ldres,
                   float alpha) {
    dim3 g((N + 127) / 128, M / 64, 1);
    k_gemm<0><<<g, 256, 0, s>>>(A, (const void*)W, bias, res, C,
                                M, N, K, lda, ldb, ldc, ldres, alpha,
                                1, 0, 0, 0, 0, 0, 0);
}

struct MambaW {
    const float *rms, *convw, *convb, *dtb, *alog, *dd;
    const __bf16 *ipw, *xpw, *dtw, *opw;
};

struct Scratch {
    float *h, *ip, *xc, *dbl, *dlt, *ys;
};

static void run_mamba(hipStream_t s, float* x, int Bn, int L,
                      const MambaW& w, const Scratch& t) {
    const int M = Bn * L;
    const long tot = (long)M * D_INNER;
    k_rmsnorm<<<M, 256, 0, s>>>(x, w.rms, t.h, D_MODEL);
    gemm_w(s, t.h, w.ipw, nullptr, nullptr, t.ip,
           M, 2 * D_INNER, D_MODEL, D_MODEL, D_MODEL, 2 * D_INNER, 0, 1.f);
    k_conv_silu<<<(unsigned)((tot + 255) / 256), 256, 0, s>>>(t.ip, w.convw, w.convb, t.xc, Bn, L);
    // xproj: weights padded to 128 rows; cols 96..127 of dbl are junk, never read
    gemm_w(s, t.xc, w.xpw, nullptr, nullptr, t.dbl,
           M, DBL_LD, D_INNER, D_INNER, D_INNER, DBL_LD, 0, 1.f);
    gemm_w(s, t.dbl, w.dtw, w.dtb, nullptr, t.dlt,
           M, D_INNER, DT_RANK, DBL_LD, DT_RANK, D_INNER, 0, 1.f);
    k_softplus<<<(unsigned)((tot + 255) / 256), 256, 0, s>>>(t.dlt, tot);
    k_scan<<<(Bn * D_INNER + 255) / 256, 256, 0, s>>>(t.dlt, t.dbl, t.xc, t.ip,
                                                      w.alog, w.dd, t.ys, Bn, L);
    gemm_w(s, t.ys, w.opw, nullptr, x, x,
           M, D_MODEL, D_INNER, D_INNER, D_INNER, D_MODEL, D_MODEL, 1.f);
}

// ---------------------------------------------------------------------------
// kernel_launch
// ---------------------------------------------------------------------------
extern "C" void kernel_launch(void* const* d_in, const int* in_sizes, int n_in,
                              void* d_out, int out_size, void* d_ws, size_t ws_size,
                              hipStream_t stream) {
    (void)in_sizes; (void)n_in; (void)out_size; (void)ws_size;

    // ---- inputs, in setup_inputs() dict order ----
    const int*   input_ids = (const int*)  d_in[0];
    const float* mels      = (const float*)d_in[1];
    const float* emb       = (const float*)d_in[2];
    const float* dec_in_w  = (const float*)d_in[3];
    const float* dec_in_b  = (const float*)d_in[4];
    const float* attn_in_w = (const float*)d_in[5];
    const float* attn_in_b = (const float*)d_in[6];
    const float* attn_out_w= (const float*)d_in[7];
    const float* attn_out_b= (const float*)d_in[8];
    const float* ln_w      = (const float*)d_in[9];
    const float* ln_b      = (const float*)d_in[10];
    const float* norm_f_w  = (const float*)d_in[11];
    const float* out_w     = (const float*)d_in[12];

    const float* e_rms  = (const float*)d_in[13];
    const float* e_ipw  = (const float*)d_in[14];
    const float* e_cw   = (const float*)d_in[15];
    const float* e_cb   = (const float*)d_in[16];
    const float* e_xpw  = (const float*)d_in[17];
    const float* e_dtw  = (const float*)d_in[18];
    const float* e_dtb  = (const float*)d_in[19];
    const float* e_al   = (const float*)d_in[20];
    const float* e_D    = (const float*)d_in[21];
    const float* e_opw  = (const float*)d_in[22];

    const float* d_rms  = (const float*)d_in[23];
    const float* d_ipw  = (const float*)d_in[24];
    const float* d_cw   = (const float*)d_in[25];
    const float* d_cb   = (const float*)d_in[26];
    const float* d_xpw  = (const float*)d_in[27];
    const float* d_dtw  = (const float*)d_in[28];
    const float* d_dtb  = (const float*)d_in[29];
    const float* d_al   = (const float*)d_in[30];
    const float* d_D    = (const float*)d_in[31];
    const float* d_opw  = (const float*)d_in[32];

    float* out = (float*)d_out;

    // ---- workspace bump allocator ----
    char* wp = (char*)d_ws;
    auto alloc = [&](size_t bytes) -> void* {
        void* r = (void*)wp;
        wp += (bytes + 255) & ~(size_t)255;
        return r;
    };

    // bf16 weight copies (padded where needed)
    const size_t nIP = (size_t)2 * D_INNER * D_MODEL;    // 4096*1024
    const size_t nXP = (size_t)DBL_LD * D_INNER;         // padded: 128*2048
    const size_t nDT = (size_t)D_INNER * DT_RANK;
    const size_t nOP = (size_t)D_MODEL * D_INNER;
    __bf16* eip = (__bf16*)alloc(N_ENC * nIP * 2);
    __bf16* exp_= (__bf16*)alloc(N_ENC * nXP * 2);
    __bf16* edt = (__bf16*)alloc(N_ENC * nDT * 2);
    __bf16* eop = (__bf16*)alloc(N_ENC * nOP * 2);
    __bf16* dip = (__bf16*)alloc(N_DEC * nIP * 2);
    __bf16* dxp = (__bf16*)alloc(N_DEC * nXP * 2);
    __bf16* ddt = (__bf16*)alloc(N_DEC * nDT * 2);
    __bf16* dop = (__bf16*)alloc(N_DEC * nOP * 2);
    __bf16* dinw= (__bf16*)alloc((size_t)D_MODEL * MEL_KP * 2);   // K padded 80->128
    __bf16* ainw= (__bf16*)alloc((size_t)N_DEC * 3 * D_MODEL * D_MODEL * 2);
    __bf16* aoutw=(__bf16*)alloc((size_t)N_DEC * D_MODEL * D_MODEL * 2);
    __bf16* outwb=(__bf16*)alloc((size_t)128 * D_MODEL * 2);      // N padded 80->128

    cvt(stream, e_ipw, eip,  (long)N_ENC * nIP);
    cvt(stream, e_dtw, edt,  (long)N_ENC * nDT);
    cvt(stream, e_opw, eop,  (long)N_ENC * nOP);
    cvt(stream, d_ipw, dip,  (long)N_DEC * nIP);
    cvt(stream, d_dtw, ddt,  (long)N_DEC * nDT);
    cvt(stream, d_opw, dop,  (long)N_DEC * nOP);
    // xproj: per-layer copy into a 128-row slot (rows 96..127 left as junk —
    // their outputs land in dbl cols 96..127 which are never read)
    for (int i = 0; i < N_ENC; ++i)
        cvt(stream, e_xpw + (size_t)i * 96 * D_INNER, exp_ + (size_t)i * nXP,
            (long)96 * D_INNER);
    for (int i = 0; i < N_DEC; ++i)
        cvt(stream, d_xpw + (size_t)i * 96 * D_INNER, dxp + (size_t)i * nXP,
            (long)96 * D_INNER);
    // dec_in: zero-pad K 80->128
    k_cvt_pad_bf16<<<(unsigned)(((long)D_MODEL * MEL_KP + 255) / 256), 256, 0, stream>>>(
        dec_in_w, dinw, D_MODEL, 80, MEL_KP);
    cvt(stream, attn_in_w, ainw, (long)N_DEC * 3 * D_MODEL * D_MODEL);
    cvt(stream, attn_out_w, aoutw, (long)N_DEC * D_MODEL * D_MODEL);
    cvt(stream, out_w, outwb, (long)80 * D_MODEL);   // rows 80..127 junk, store-guarded

    // f32 activation buffers (sized for the larger decoder path)
    float* xenc = (float*)alloc((size_t)M_ENC * D_MODEL * 4);   // encoder stream / enc_out
    float* ydec = (float*)alloc((size_t)M_DEC * D_MODEL * 4);   // decoder stream
    float* melp = (float*)alloc((size_t)M_DEC * MEL_KP * 4);    // zero-padded mels
    Scratch sc;
    sc.h   = (float*)alloc((size_t)M_DEC * D_MODEL * 4);
    sc.ip  = (float*)alloc((size_t)M_DEC * 2 * D_INNER * 4);
    sc.xc  = (float*)alloc((size_t)M_DEC * D_INNER * 4);
    sc.dbl = (float*)alloc((size_t)M_DEC * DBL_LD * 4);
    sc.dlt = (float*)alloc((size_t)M_DEC * D_INNER * 4);
    sc.ys  = (float*)alloc((size_t)M_DEC * D_INNER * 4);
    float* qb  = (float*)alloc((size_t)M_DEC * D_MODEL * 4);
    float* kvb = (float*)alloc((size_t)M_ENC * 2 * D_MODEL * 4);
    float* scb = (float*)alloc((size_t)BATCH * N_HEADS * L_MEL * L_TXT * 4);
    float* ob  = (float*)alloc((size_t)M_DEC * D_MODEL * 4);
    float* tb  = (float*)alloc((size_t)M_DEC * D_MODEL * 4);

    // ---- encoder ----
    k_embed<<<(unsigned)(((long)M_ENC * D_MODEL + 255) / 256), 256, 0, stream>>>(
        input_ids, emb, xenc, M_ENC, D_MODEL);
    for (int i = 0; i < N_ENC; ++i) {
        MambaW w;
        w.rms   = e_rms + (size_t)i * D_MODEL;
        w.convw = e_cw  + (size_t)i * D_INNER * D_CONV;
        w.convb = e_cb  + (size_t)i * D_INNER;
        w.dtb   = e_dtb + (size_t)i * D_INNER;
        w.alog  = e_al  + (size_t)i * D_INNER * D_STATE;
        w.dd    = e_D   + (size_t)i * D_INNER;
        w.ipw   = eip  + (size_t)i * nIP;
        w.xpw   = exp_ + (size_t)i * nXP;
        w.dtw   = edt  + (size_t)i * nDT;
        w.opw   = eop  + (size_t)i * nOP;
        run_mamba(stream, xenc, BATCH, L_TXT, w, sc);
    }
    // xenc now holds enc_out [2048, 1024]

    // ---- decoder input projection: y = mels @ dec_in_w.T + b (K padded) ----
    k_pad_f32<<<(unsigned)(((long)M_DEC * MEL_KP + 255) / 256), 256, 0, stream>>>(
        mels, melp, M_DEC, 80, MEL_KP);
    gemm_w(stream, melp, dinw, dec_in_b, nullptr, ydec,
           M_DEC, D_MODEL, MEL_KP, MEL_KP, MEL_KP, D_MODEL, 0, 1.f);

    // ---- decoder layers ----
    for (int i = 0; i < N_DEC; ++i) {
        MambaW w;
        w.rms   = d_rms + (size_t)i * D_MODEL;
        w.convw = d_cw  + (size_t)i * D_INNER * D_CONV;
        w.convb = d_cb  + (size_t)i * D_INNER;
        w.dtb   = d_dtb + (size_t)i * D_INNER;
        w.alog  = d_al  + (size_t)i * D_INNER * D_STATE;
        w.dd    = d_D   + (size_t)i * D_INNER;
        w.ipw   = dip + (size_t)i * nIP;
        w.xpw   = dxp + (size_t)i * nXP;
        w.dtw   = ddt + (size_t)i * nDT;
        w.opw   = dop + (size_t)i * nOP;
        run_mamba(stream, ydec, BATCH, L_MEL, w, sc);

        // cross attention against enc_out
        const __bf16* Wqkv = ainw + (size_t)i * 3 * D_MODEL * D_MODEL;
        const float*  bqkv = attn_in_b + (size_t)i * 3 * D_MODEL;
        // Q = y @ Wq^T + bq                            [8192, 1024]
        gemm_w(stream, ydec, Wqkv, bqkv, nullptr, qb,
               M_DEC, D_MODEL, D_MODEL, D_MODEL, D_MODEL, D_MODEL, 0, 1.f);
        // KV = enc_out @ [Wk;Wv]^T + [bk;bv]           [2048, 2048]
        gemm_w(stream, xenc, Wqkv + (size_t)D_MODEL * D_MODEL, bqkv + D_MODEL,
               nullptr, kvb,
               M_ENC, 2 * D_MODEL, D_MODEL, D_MODEL, D_MODEL, 2 * D_MODEL, 0, 1.f);
        // scores[b,h] = Q_h K_h^T / sqrt(dh)           batched over 32 (b,h)
        {
            dim3 g(L_TXT / 128, L_MEL / 64, BATCH * N_HEADS);
            k_gemm<1><<<g, 256, 0, stream>>>(
                qb, (const void*)kvb, nullptr, nullptr, scb,
                L_MEL, L_TXT, D_HEAD, D_MODEL, 2 * D_MODEL, L_TXT, 0, 0.0625f,
                N_HEADS,
                (long)L_MEL * D_MODEL, (long)D_HEAD,          // A strides (b,h)
                (long)L_TXT * 2 * D_MODEL, (long)D_HEAD,      // B strides
                (long)N_HEADS * L_MEL * L_TXT, (long)L_MEL * L_TXT); // C strides
        }
        k_softmax256<<<BATCH * N_HEADS * L_MEL, 256, 0, stream>>>(scb);
        // O_h = P V_h                                   batched over 32 (b,h)
        {
            dim3 g(D_HEAD / 128, L_MEL / 64, BATCH * N_HEADS);
            k_gemm<2><<<g, 256, 0, stream>>>(
                scb, (const void*)(kvb + D_MODEL), nullptr, nullptr, ob,
                L_MEL, D_HEAD, L_TXT, L_TXT, 2 * D_MODEL, D_MODEL, 0, 1.f,
                N_HEADS,
                (long)N_HEADS * L_MEL * L_TXT, (long)L_MEL * L_TXT,
                (long)L_TXT * 2 * D_MODEL, (long)D_HEAD,
                (long)L_MEL * D_MODEL, (long)D_HEAD);
        }
        // out proj + residual, then LayerNorm back into ydec
        gemm_w(stream, ob, aoutw + (size_t)i * D_MODEL * D_MODEL,
               attn_out_b + (size_t)i * D_MODEL, ydec, tb,
               M_DEC, D_MODEL, D_MODEL, D_MODEL, D_MODEL, D_MODEL, D_MODEL, 1.f);
        k_layernorm<<<M_DEC, 256, 0, stream>>>(tb, ln_w + (size_t)i * D_MODEL,
                                               ln_b + (size_t)i * D_MODEL, ydec, D_MODEL);
    }

    // ---- final norm + output projection to 80 mels ----
    k_rmsnorm<<<M_DEC, 256, 0, stream>>>(ydec, norm_f_w, sc.h, D_MODEL);
    gemm_w(stream, sc.h, outwb, nullptr, nullptr, out,
           M_DEC, 80, D_MODEL, D_MODEL, D_MODEL, 80, 0, 1.f);
}
